// CapsuleNetwork_10892037063179
// MI455X (gfx1250) — matrix operand order repaired and checked by reading, not compiled, not run on hardware
//
#include <hip/hip_runtime.h>
#include <math.h>

// ---------------------------------------------------------------------------
// CDNA5 (gfx1250) wave32 WMMA types & helpers
// ---------------------------------------------------------------------------
typedef __attribute__((ext_vector_type(16))) __bf16 v16bf;
typedef __attribute__((ext_vector_type(8)))  float  v8f;

union BF16Frag {
    v16bf v;
    unsigned short u[16];
    uint4 q[2];
};

__device__ __forceinline__ unsigned short f2bf(float f) {
    union { float f; unsigned u; } cv; cv.f = f;
    unsigned x = cv.u;
    x += 0x7FFFu + ((x >> 16) & 1u);   // round-to-nearest-even
    return (unsigned short)(x >> 16);
}

// --------------------------- async copy to LDS ------------------------------
#define AS1 __attribute__((address_space(1)))
#define AS3 __attribute__((address_space(3)))

typedef int vint4 __attribute__((vector_size(16)));   // 16-byte int vector

#if __has_builtin(__builtin_amdgcn_global_load_async_to_lds_b128)
#define HAS_ASYNC 1
#else
#define HAS_ASYNC 0
#endif

// 16-byte global -> LDS copy; async (ASYNCcnt-tracked) when available.
__device__ __forceinline__ void copy16(unsigned short* l, const unsigned short* g) {
#if HAS_ASYNC
    __builtin_amdgcn_global_load_async_to_lds_b128(
        (AS1 vint4*)(g), (AS3 vint4*)(l), 0, 0);
#else
    *(uint4*)l = *(const uint4*)g;
#endif
}

__device__ __forceinline__ void async_wait() {
#if HAS_ASYNC
#if __has_builtin(__builtin_amdgcn_s_wait_asynccnt)
    __builtin_amdgcn_s_wait_asynccnt(0);
#else
    asm volatile("s_wait_asynccnt 0x0" ::: "memory");
#endif
#endif
}

// ISA 7.12.2 fragment index maps (wave32, 16-bit A 16x32 / B 32x16 / f32 C 16x16):
//   A: lane half h, element e:  m = lane&15, k = (e<8 ? e : e+8) + 8*h
//   B: lane half h, element e:  n = lane&15, k = (e<8 ? e : e+8) + 8*h
//   C: element e:               m = e + 8*h, n = lane&15

// ---------------------------------------------------------------------------
// K0: caps1 weight transform [co=256][ci=256][r=81] f32 ->
//     fragment-ready bf16 layout [r][chunk=4][co=256][ci64]  (linear-copyable)
// ---------------------------------------------------------------------------
__global__ void __launch_bounds__(256) w1_transform_kernel(
    const float* __restrict__ W, unsigned short* __restrict__ Wbf) {
    int idx = blockIdx.x * 256 + threadIdx.x;
    if (idx >= 81 * 256 * 256) return;
    int cil   = idx & 63;
    int co    = (idx >> 6) & 255;
    int chunk = (idx >> 14) & 3;
    int r     = idx >> 16;
    int ci    = chunk * 64 + cil;
    Wbf[idx] = f2bf(W[(co * 256 + ci) * 81 + r]);
}

// ---------------------------------------------------------------------------
// K1: conv1 (9x9, stride1) + bias + ReLU as gather-WMMA implicit GEMM.
//     M = 256*20*20 = 102400, N = 256, K = 81 (3 k-steps of 32).
//     Output stored NHWC bf16: Hbf[(b*400 + y*20 + x)*256 + co].
// ---------------------------------------------------------------------------
__global__ void __launch_bounds__(128) conv1_kernel(
    const float* __restrict__ X, const float* __restrict__ CW,
    const float* __restrict__ CB, unsigned short* __restrict__ Hbf) {
    const int lane = threadIdx.x & 31;
    const int wave = threadIdx.x >> 5;
    const int tile = blockIdx.x * 4 + wave;
    const int tile_n = tile & 15;          // 16 N-tiles
    const int tile_m = tile >> 4;          // 6400 M-tiles
    const int half = lane >> 4;
    const int lmod = lane & 15;

    const int gm = tile_m * 16 + lmod;     // A row for this lane
    const int bb = gm / 400;
    const int p  = gm % 400;
    const int yy = p / 20, xx = p % 20;
    const int co = tile_n * 16 + lmod;     // B col for this lane

    v8f acc = {0.f,0.f,0.f,0.f,0.f,0.f,0.f,0.f};
    for (int ks = 0; ks < 3; ++ks) {
        BF16Frag A, B;
#pragma unroll
        for (int e = 0; e < 16; ++e) {
            int ka = ((e < 8) ? e : e + 8) + half * 8 + ks * 32;
            float av = 0.f;
            if (ka < 81) {
                int ky = ka / 9, kx = ka % 9;
                av = X[bb * 784 + (yy + ky) * 28 + (xx + kx)];
            }
            A.u[e] = f2bf(av);
            int kb = ((e < 8) ? e : e + 8) + half * 8 + ks * 32;
            float bv = (kb < 81) ? CW[co * 81 + kb] : 0.f;
            B.u[e] = f2bf(bv);
        }
        acc = __builtin_amdgcn_wmma_f32_16x16x32_bf16(
            false, A.v, false, B.v, (short)0, acc, false, false);
    }
#pragma unroll
    for (int e = 0; e < 8; ++e) {
        int m  = e + 8 * half;
        int g  = tile_m * 16 + m;
        int b2 = g / 400;
        int p2 = g % 400;
        float v = acc[e] + CB[co];
        v = v > 0.f ? v : 0.f;
        Hbf[(b2 * 400 + p2) * 256 + co] = f2bf(v);
    }
}

// ---------------------------------------------------------------------------
// K2: caps1 conv (9x9 stride2, 256->256ch) implicit GEMM, async-LDS staged.
//     M = 256*36 = 9216, N = 256, K = 81*256 = 20736.
//     Block: 256 threads (8 waves), M-tile 32, N full 256; wave w owns
//     n-subtiles {2w,2w+1} x m-subtiles {0,1} = 4 accumulators.
//     A tile: sA[m][ci] (rows linear in Hbf NHWC -> 16B async copies).
//     B tile: sB[co][ci64 (+pad to 72)] staged from the pre-swizzled
//             W1bf[r][chunk][co][ci64] global layout -> 16B async copies,
//             fragment reads stay aligned ds_load_b128.
// ---------------------------------------------------------------------------
__global__ void __launch_bounds__(256) caps1_kernel(
    const unsigned short* __restrict__ Hbf, const unsigned short* __restrict__ Wbf,
    const float* __restrict__ B1, float* __restrict__ U) {
    __shared__ __align__(16) unsigned short sA[32 * 256];   // [m][ci]        16 KB
    __shared__ __align__(16) unsigned short sB[256 * 72];   // [co][ci64+pad] 36 KB
    const int t    = threadIdx.x;
    const int lane = t & 31;
    const int wave = t >> 5;
    const int half = lane >> 4;
    const int lmod = lane & 15;
    const int blockM = blockIdx.x * 32;

    // A staging assignment: 8 threads per m-row, 4 x 16B segments each
    const int arow = t >> 3;           // 0..31
    const int aseg = t & 7;            // 0..7
    const int gmA  = blockM + arow;
    const int bbA  = gmA / 36, posA = gmA % 36;
    const int oyA  = posA / 6, oxA = posA % 6;

    const v8f zero = {0.f,0.f,0.f,0.f,0.f,0.f,0.f,0.f};
    v8f acc[2][2] = {{zero, zero}, {zero, zero}};

    for (int r = 0; r < 81; ++r) {
        const int ky = r / 9, kx = r % 9;
        __syncthreads();                              // prior chunk compute done
        // Prefetch next r's weight block toward L2 (global_prefetch_b8)
        if (r + 1 < 81)
            __builtin_prefetch(Wbf + (((size_t)(r + 1) * 4) * 256 + t) * 64, 0, 1);
        // Stage A: row arow, segments aseg, aseg+8, aseg+16, aseg+24 (16B each)
        {
            const unsigned short* gRow =
                Hbf + (size_t)((bbA * 400 + (2 * oyA + ky) * 20 + (2 * oxA + kx))) * 256;
#pragma unroll
            for (int j = 0; j < 4; ++j) {
                int s = aseg + 8 * j;                 // 16B segment index (8 u16)
                copy16(&sA[arow * 256 + s * 8], gRow + s * 8);
            }
        }
        for (int chunk = 0; chunk < 4; ++chunk) {
            if (chunk) __syncthreads();
            // Stage B: thread t = co row; 8 x 16B copies from swizzled weights
            {
                const unsigned short* gB =
                    Wbf + (((size_t)(r * 4 + chunk) * 256 + t) * 64);
#pragma unroll
                for (int i = 0; i < 8; ++i)
                    copy16(&sB[t * 72 + i * 8], gB + i * 8);
            }
            async_wait();
            __syncthreads();                          // all staged & visible
#pragma unroll
            for (int ks = 0; ks < 2; ++ks) {
                BF16Frag A[2], Bf[2];
#pragma unroll
                for (int ms = 0; ms < 2; ++ms) {
                    const unsigned short* ap =
                        &sA[(ms * 16 + lmod) * 256 + chunk * 64 + ks * 32 + half * 8];
                    A[ms].q[0] = *(const uint4*)(ap);
                    A[ms].q[1] = *(const uint4*)(ap + 16);
                }
#pragma unroll
                for (int ns = 0; ns < 2; ++ns) {
                    int co = (wave * 2 + ns) * 16 + lmod;
                    const unsigned short* bp = &sB[co * 72 + ks * 32 + half * 8];
                    Bf[ns].q[0] = *(const uint4*)(bp);
                    Bf[ns].q[1] = *(const uint4*)(bp + 16);
                }
#pragma unroll
                for (int ms = 0; ms < 2; ++ms)
#pragma unroll
                    for (int ns = 0; ns < 2; ++ns)
                        acc[ms][ns] = __builtin_amdgcn_wmma_f32_16x16x32_bf16(
                            false, A[ms].v, false, Bf[ns].v, (short)0,
                            acc[ms][ns], false, false);
            }
        }
    }
    // Epilogue: reorder into [b][n = oy*192 + caps*6 + ox][ch]
#pragma unroll
    for (int ms = 0; ms < 2; ++ms) {
#pragma unroll
        for (int ns = 0; ns < 2; ++ns) {
            int co = (wave * 2 + ns) * 16 + lmod;
            int caps = co >> 3, ch = co & 7;
#pragma unroll
            for (int e = 0; e < 8; ++e) {
                int gm = blockM + ms * 16 + e + 8 * half;
                int bb = gm / 36, pos = gm % 36;
                int oy = pos / 6, ox = pos % 6;
                int n = oy * 192 + caps * 6 + ox;
                U[((size_t)bb * 1152 + n) * 8 + ch] = acc[ms][ns][e] + B1[co];
            }
        }
    }
}

// ---------------------------------------------------------------------------
// K3: squash over the 8-dim (in place)
// ---------------------------------------------------------------------------
__global__ void __launch_bounds__(256) squash_u_kernel(float* __restrict__ U) {
    int row = blockIdx.x * 256 + threadIdx.x;
    if (row >= 256 * 1152) return;
    float* p = U + (size_t)row * 8;
    float s[8], sq = 0.f;
#pragma unroll
    for (int i = 0; i < 8; ++i) { s[i] = p[i]; sq += s[i] * s[i]; }
    float scale = (sq / (1.f + sq)) * rsqrtf(sq + 1e-8f);
#pragma unroll
    for (int i = 0; i < 8; ++i) p[i] = s[i] * scale;
}

// ---------------------------------------------------------------------------
// K4: fused x_hat + 3-iteration dynamic routing. One block per (class,batch).
//     288 threads: thread t owns n = 4t..4t+3, x_hat kept in registers.
// ---------------------------------------------------------------------------
__global__ void __launch_bounds__(288) routing_kernel(
    const float* __restrict__ U, const float* __restrict__ W2,
    float* __restrict__ Vout) {
    const int t  = threadIdx.x;
    const int c  = blockIdx.x % 10;
    const int bb = blockIdx.x / 10;
    __shared__ float red[288];
    __shared__ float s_sh[16];

    float xh[4][16];
    float beta[4];
#pragma unroll
    for (int j = 0; j < 4; ++j) {
        int n = t * 4 + j;
        const float* up = U + ((size_t)bb * 1152 + n) * 8;
        const float* wp = W2 + ((size_t)(c * 1152 + n) * 8) * 16;
        float uv[8];
#pragma unroll
        for (int i = 0; i < 8; ++i) uv[i] = up[i];
#pragma unroll
        for (int o = 0; o < 16; ++o) {
            float a = 0.f;
#pragma unroll
            for (int i = 0; i < 8; ++i) a += uv[i] * wp[i * 16 + o];
            xh[j][o] = a;
        }
        beta[j] = 0.f;
    }

    float v[16];
    for (int it = 0; it < 3; ++it) {
        float lmax = fmaxf(fmaxf(beta[0], beta[1]), fmaxf(beta[2], beta[3]));
        red[t] = lmax;
        __syncthreads();
        if (t == 0) { float m = red[0]; for (int k = 1; k < 288; ++k) m = fmaxf(m, red[k]); red[0] = m; }
        __syncthreads();
        float gmax = red[0];
        __syncthreads();
        float e[4], lsum = 0.f;
#pragma unroll
        for (int j = 0; j < 4; ++j) { e[j] = __expf(beta[j] - gmax); lsum += e[j]; }
        red[t] = lsum;
        __syncthreads();
        if (t == 0) { float s = 0.f; for (int k = 0; k < 288; ++k) s += red[k]; red[0] = s; }
        __syncthreads();
        float inv = 1.f / red[0];
        if (t < 16) s_sh[t] = 0.f;
        __syncthreads();
#pragma unroll
        for (int o = 0; o < 16; ++o) {
            float a = 0.f;
#pragma unroll
            for (int j = 0; j < 4; ++j) a += e[j] * xh[j][o];
            atomicAdd(&s_sh[o], a * inv);
        }
        __syncthreads();
        float sq = 0.f;
#pragma unroll
        for (int o = 0; o < 16; ++o) { float sv = s_sh[o]; sq += sv * sv; }
        float scale = (sq / (1.f + sq)) * rsqrtf(sq + 1e-8f);
#pragma unroll
        for (int o = 0; o < 16; ++o) v[o] = s_sh[o] * scale;
        if (it < 2) {
#pragma unroll
            for (int j = 0; j < 4; ++j) {
                float d = 0.f;
#pragma unroll
                for (int o = 0; o < 16; ++o) d += xh[j][o] * v[o];
                beta[j] += d;
            }
        }
    }
    if (t < 16) Vout[((size_t)bb * 10 + c) * 16 + t] = v[t];
}

// ---------------------------------------------------------------------------
// K5: class scores -> softmax probs (output) + argmax one-hot masking
// ---------------------------------------------------------------------------
__global__ void __launch_bounds__(256) mask_kernel(
    const float* __restrict__ Vout, float* __restrict__ probs,
    float* __restrict__ Masked) {
    const int b = threadIdx.x;
    const float* vp = Vout + (size_t)b * 160;
    float score[10];
#pragma unroll
    for (int c = 0; c < 10; ++c) {
        float sq = 0.f;
#pragma unroll
        for (int o = 0; o < 16; ++o) { float x = vp[c * 16 + o]; sq += x * x; }
        score[c] = sqrtf(sq + 1e-8f);
    }
    float m = score[0];
#pragma unroll
    for (int c = 1; c < 10; ++c) m = fmaxf(m, score[c]);
    float pe[10], sum = 0.f;
#pragma unroll
    for (int c = 0; c < 10; ++c) { pe[c] = __expf(score[c] - m); sum += pe[c]; }
    float inv = 1.f / sum;
    int best = 0; float bp = pe[0];
#pragma unroll
    for (int c = 1; c < 10; ++c) if (pe[c] > bp) { bp = pe[c]; best = c; }
#pragma unroll
    for (int c = 0; c < 10; ++c) probs[b * 10 + c] = pe[c] * inv;
#pragma unroll
    for (int c = 0; c < 10; ++c)
#pragma unroll
        for (int o = 0; o < 16; ++o)
            Masked[(size_t)b * 160 + c * 16 + o] = (c == best) ? vp[c * 16 + o] : 0.f;
}

// ---------------------------------------------------------------------------
// K6-K8: decoder FC layers as gather-WMMA bf16 GEMM + bias + activation
//        C[M,N] = act(A[M,K] @ B[K,N] + bias);  K % 32 == 0, N % 16 == 0
// ---------------------------------------------------------------------------
__global__ void __launch_bounds__(128) fc_kernel(
    const float* __restrict__ A, const float* __restrict__ Bm,
    const float* __restrict__ bias, float* __restrict__ C,
    int M, int N, int K, int act /*0=relu 1=sigmoid*/) {
    const int lane = threadIdx.x & 31;
    const int wave = threadIdx.x >> 5;
    const int tile = blockIdx.x * 4 + wave;
    const int ntiles = N >> 4;
    const int tile_n = tile % ntiles;
    const int tile_m = tile / ntiles;
    const int half = lane >> 4;
    const int lmod = lane & 15;
    const int gm = tile_m * 16 + lmod;
    const int co = tile_n * 16 + lmod;
    if (tile_m * 16 >= M) return;

    v8f acc = {0.f,0.f,0.f,0.f,0.f,0.f,0.f,0.f};
    for (int ks = 0; ks < (K >> 5); ++ks) {
        BF16Frag Af, Bf;
#pragma unroll
        for (int e = 0; e < 16; ++e) {
            int ka = ((e < 8) ? e : e + 8) + half * 8 + ks * 32;
            Af.u[e] = f2bf(A[(size_t)gm * K + ka]);
            int kb = ((e < 8) ? e : e + 8) + half * 8 + ks * 32;
            Bf.u[e] = f2bf(Bm[(size_t)kb * N + co]);
        }
        acc = __builtin_amdgcn_wmma_f32_16x16x32_bf16(
            false, Af.v, false, Bf.v, (short)0, acc, false, false);
    }
#pragma unroll
    for (int e = 0; e < 8; ++e) {
        int m = tile_m * 16 + e + 8 * half;
        float v = acc[e] + bias[co];
        if (act == 0) v = v > 0.f ? v : 0.f;
        else          v = 1.f / (1.f + __expf(-v));
        C[(size_t)m * N + co] = v;
    }
}

// ---------------------------------------------------------------------------
// Host launcher
// ---------------------------------------------------------------------------
extern "C" void kernel_launch(void* const* d_in, const int* in_sizes, int n_in,
                              void* d_out, int out_size, void* d_ws, size_t ws_size,
                              hipStream_t stream) {
    const float* X   = (const float*)d_in[0];   // [256,1,28,28]
    const float* CW  = (const float*)d_in[1];   // [256,1,9,9]
    const float* CB  = (const float*)d_in[2];   // [256]
    const float* C1W = (const float*)d_in[3];   // [32,8,256,9,9]
    const float* C1B = (const float*)d_in[4];   // [32,8]
    const float* C2W = (const float*)d_in[5];   // [10,1152,8,16]
    const float* DW1 = (const float*)d_in[6];   // [160,512]
    const float* DB1 = (const float*)d_in[7];
    const float* DW2 = (const float*)d_in[8];   // [512,1024]
    const float* DB2 = (const float*)d_in[9];
    const float* DW3 = (const float*)d_in[10];  // [1024,784]
    const float* DB3 = (const float*)d_in[11];

    char* ws = (char*)d_ws;
    size_t off = 0;
    auto alloc = [&](size_t bytes) -> void* {
        void* p = ws + off;
        off = (off + bytes + 255) & ~(size_t)255;
        return p;
    };
    unsigned short* W1bf = (unsigned short*)alloc(81ull * 256 * 256 * 2);   // 10.6 MB
    unsigned short* Hbf  = (unsigned short*)alloc(256ull * 400 * 256 * 2);  // 52.4 MB
    float* U    = (float*)alloc(256ull * 1152 * 8 * 4);                     // 9.4 MB
    float* Vout = (float*)alloc(256ull * 160 * 4);
    float* Msk  = (float*)alloc(256ull * 160 * 4);
    float* D1   = (float*)alloc(256ull * 512 * 4);
    float* D2   = (float*)alloc(256ull * 1024 * 4);

    float* probs = (float*)d_out;          // [256,10]
    float* recon = (float*)d_out + 2560;   // [256,784]

    w1_transform_kernel<<<20736, 256, 0, stream>>>(C1W, W1bf);
    conv1_kernel<<<25600, 128, 0, stream>>>(X, CW, CB, Hbf);
    caps1_kernel<<<288, 256, 0, stream>>>(Hbf, W1bf, C1B, U);
    squash_u_kernel<<<1152, 256, 0, stream>>>(U);
    routing_kernel<<<2560, 288, 0, stream>>>(U, C2W, Vout);
    mask_kernel<<<1, 256, 0, stream>>>(Vout, probs, Msk);
    fc_kernel<<<128, 128, 0, stream>>>(Msk, DW1, DB1, D1, 256, 512, 160, 0);
    fc_kernel<<<256, 128, 0, stream>>>(D1, DW2, DB2, D2, 256, 1024, 512, 0);
    fc_kernel<<<196, 128, 0, stream>>>(D2, DW3, DB3, recon, 256, 784, 1024, 1);
}
